// MSDeformAttnTransformerEncoderLayer_9680856285298
// MI455X (gfx1250) — matrix operand order repaired and checked by reading, not compiled, not run on hardware
//
#include <hip/hip_runtime.h>

// ---------------------------------------------------------------------------
// MSDeformAttn encoder layer for MI455X (gfx1250, wave32, WMMA 16x16x32 f16)
// ---------------------------------------------------------------------------
// Fixed problem sizes from the reference:
//   D=256, NH=8, NP=4, NL=4, DH=32, MLP=1024, B=2
//   LQ = 128^2 + 64^2 + 32^2 + 16^2 = 21760, M = B*LQ = 43520
//   level starts = {0, 16384, 20480, 21504}, level sizes = {128,64,32,16}^2
// M is a multiple of 128, all GEMM N are multiples of 64, K multiples of 32,
// so the WMMA GEMM needs no edge handling and EXEC stays all-ones.

typedef __attribute__((ext_vector_type(16))) _Float16 v16h;
typedef __attribute__((ext_vector_type(8)))  _Float16 v8h;
typedef __attribute__((ext_vector_type(8)))  float    v8f;

// ---------------------------------------------------------------------------
// Weight convert + transpose: W[K][N] (f32, row-major) -> Wt[N][K] (f16).
// Wt rows are contiguous in K, which suits the GEMM's B-matrix
// fragment loads (each lane = one N column, 16 contiguous K halves).
// ---------------------------------------------------------------------------
__global__ void convert_w_kernel(const float* __restrict__ W,
                                 _Float16* __restrict__ Wt,
                                 int K, int N)
{
    int t = blockIdx.x * blockDim.x + threadIdx.x;
    if (t >= K * N) return;
    int n = t / K;
    int k = t - n * K;
    Wt[(size_t)n * K + k] = (_Float16)W[(size_t)k * N + n];
}

// ---------------------------------------------------------------------------
// LayerNorm over 256 channels: one wave (32 lanes) per row, 8 elems/lane.
// Writes f16 LN output; optionally also f16 (LN + pos) for the query path.
// ---------------------------------------------------------------------------
__global__ __launch_bounds__(256) void ln_kernel(const float* __restrict__ x,
                                                 const float* __restrict__ pos,
                                                 const float* __restrict__ g,
                                                 const float* __restrict__ beta,
                                                 _Float16* __restrict__ out_ln,
                                                 _Float16* __restrict__ out_q,
                                                 int Mrows)
{
    const int wave = threadIdx.x >> 5;
    const int lane = threadIdx.x & 31;
    const int row  = blockIdx.x * 8 + wave;
    if (row >= Mrows) return;

    const float* xr = x + (size_t)row * 256;
    float v[8];
    float s = 0.0f;
#pragma unroll
    for (int i = 0; i < 8; ++i) { v[i] = xr[lane + i * 32]; s += v[i]; }
#pragma unroll
    for (int m = 16; m > 0; m >>= 1) s += __shfl_xor(s, m, 32);
    const float mu = s * (1.0f / 256.0f);

    float var = 0.0f;
#pragma unroll
    for (int i = 0; i < 8; ++i) { float d = v[i] - mu; var += d * d; }
#pragma unroll
    for (int m = 16; m > 0; m >>= 1) var += __shfl_xor(var, m, 32);
    const float rstd = rsqrtf(var * (1.0f / 256.0f) + 1e-6f);

#pragma unroll
    for (int i = 0; i < 8; ++i) {
        const int c = lane + i * 32;
        const float y = (v[i] - mu) * rstd * g[c] + beta[c];
        out_ln[(size_t)row * 256 + c] = (_Float16)y;
        if (out_q)
            out_q[(size_t)row * 256 + c] = (_Float16)(y + pos[(size_t)row * 256 + c]);
    }
}

// ---------------------------------------------------------------------------
// WMMA GEMM: C[M][N] = act(A[M][K] @ Wt[N][K]^T + bias) (+ residual)
//   A:  f16 row-major activations
//   Wt: f16 transposed weights (N-major), resident in L2 (<=0.5 MB each)
//   outputs: optional f32 (Cf) and/or f16 (Ch)
// Workgroup = 256 threads = 8 waves -> 128(M) x 64(N) tile.
// Each wave owns a 32x32 macro-tile = 2x2 v_wmma_f32_16x16x32_f16 frags.
// Requires: M%128==0, N%64==0, K%32==0 (true for all launches here).
// ---------------------------------------------------------------------------
__device__ __forceinline__ void store_tile(v8f acc, int mBase, int nCol, int N,
                                           const float* __restrict__ bias,
                                           const float* __restrict__ residual,
                                           float* __restrict__ Cf,
                                           _Float16* __restrict__ Ch,
                                           int doRelu)
{
    const float bv = bias ? bias[nCol] : 0.0f;
#pragma unroll
    for (int r = 0; r < 8; ++r) {
        float v = acc[r] + bv;
        if (doRelu) v = fmaxf(v, 0.0f);
        const size_t idx = (size_t)(mBase + r) * N + nCol;
        if (residual) v += residual[idx];
        if (Cf) Cf[idx] = v;
        if (Ch) Ch[idx] = (_Float16)v;
    }
}

__global__ __launch_bounds__(256) void gemm_wmma_kernel(
    const _Float16* __restrict__ A,
    const _Float16* __restrict__ Wt,
    const float*    __restrict__ bias,
    const float*    __restrict__ residual,
    float*          __restrict__ Cf,
    _Float16*       __restrict__ Ch,
    int M, int N, int K, int doRelu)
{
    const int lane  = threadIdx.x & 31;
    const int wave  = threadIdx.x >> 5;   // 0..7
    const int waveM = wave >> 1;          // 0..3
    const int waveN = wave & 1;           // 0..1
    const int colBlocks = N >> 6;
    const int bm = blockIdx.x / colBlocks;
    const int bn = blockIdx.x - bm * colBlocks;
    const int m0 = bm * 128 + waveM * 32;
    const int n0 = bn * 64  + waveN * 32;
    const int r15 = lane & 15;
    const int hi  = lane >> 4;

    // A fragment layout (16-bit A, 16x32): lanes 0-15 hold K k..k+15 contiguous,
    // lanes 16-31 hold K k+16..k+31 of row (m0 + lane&15).
    const _Float16* aBase0 = A + (size_t)(m0 + r15) * K + hi * 16;
    const _Float16* aBase1 = aBase0 + (size_t)16 * K;
    // B fragment layout (16-bit B, 32x16): lane = column n0+(lane&15);
    // lanes 0-15 hold K k..k+15 contiguous, lanes 16-31 hold K k+16..k+31.
    const _Float16* bBase0 = Wt + (size_t)(n0 + r15) * K + hi * 16;
    const _Float16* bBase1 = bBase0 + (size_t)16 * K;

    v8f acc00 = {}, acc01 = {}, acc10 = {}, acc11 = {};

    for (int k = 0; k < K; k += 32) {
        union { v16h v; v8h h[2]; } a0u, a1u;
        a0u.h[0] = *(const v8h*)(aBase0 + k);
        a0u.h[1] = *(const v8h*)(aBase0 + k + 8);
        a1u.h[0] = *(const v8h*)(aBase1 + k);
        a1u.h[1] = *(const v8h*)(aBase1 + k + 8);
        const v16h b0 = *(const v16h*)(bBase0 + k);
        const v16h b1 = *(const v16h*)(bBase1 + k);

        acc00 = __builtin_amdgcn_wmma_f32_16x16x32_f16(false, a0u.v, false, b0,
                                                       (short)0, acc00, false, false);
        acc01 = __builtin_amdgcn_wmma_f32_16x16x32_f16(false, a0u.v, false, b1,
                                                       (short)0, acc01, false, false);
        acc10 = __builtin_amdgcn_wmma_f32_16x16x32_f16(false, a1u.v, false, b0,
                                                       (short)0, acc10, false, false);
        acc11 = __builtin_amdgcn_wmma_f32_16x16x32_f16(false, a1u.v, false, b1,
                                                       (short)0, acc11, false, false);
    }

    // C/D layout: VGPR r, lanes 0-15 -> M = base + r, lanes 16-31 -> M = base+8+r.
    const int mB0 = m0 + hi * 8;
    const int mB1 = m0 + 16 + hi * 8;
    const int nC0 = n0 + r15;
    const int nC1 = n0 + 16 + r15;
    store_tile(acc00, mB0, nC0, N, bias, residual, Cf, Ch, doRelu);
    store_tile(acc01, mB0, nC1, N, bias, residual, Cf, Ch, doRelu);
    store_tile(acc10, mB1, nC0, N, bias, residual, Cf, Ch, doRelu);
    store_tile(acc11, mB1, nC1, N, bias, residual, Cf, Ch, doRelu);
}

// ---------------------------------------------------------------------------
// Softmax over the 16 (NL*NP) attention logits of each (b,q,h), in place.
// ---------------------------------------------------------------------------
__global__ void softmax16_kernel(float* __restrict__ aw, int total)
{
    int t = blockIdx.x * blockDim.x + threadIdx.x;
    if (t >= total) return;
    float* p = aw + (size_t)t * 16;
    float v[16];
    float mx = -3.402823466e+38f;
#pragma unroll
    for (int i = 0; i < 16; ++i) { v[i] = p[i]; mx = fmaxf(mx, v[i]); }
    float s = 0.0f;
#pragma unroll
    for (int i = 0; i < 16; ++i) { v[i] = __expf(v[i] - mx); s += v[i]; }
    const float inv = 1.0f / s;
#pragma unroll
    for (int i = 0; i < 16; ++i) p[i] = v[i] * inv;
}

// ---------------------------------------------------------------------------
// Deformable bilinear sampling. One wave per (b,q,h); lane = channel d (DH=32).
// The per-sample scalar math is wave-uniform; the 4 corner gathers per point
// are 128-byte coalesced loads across the 32 channels.
// Writes the head-concatenated result directly as f16 for the w_out GEMM.
// ---------------------------------------------------------------------------
__global__ __launch_bounds__(256) void deform_sample_kernel(
    const float* __restrict__ val,   // [B, LQ, NH, DH] f32
    const float* __restrict__ off,   // [B, LQ, NH*NL*NP*2] f32
    const float* __restrict__ aw,    // [B, LQ, NH*NL*NP] f32 (softmaxed)
    const float* __restrict__ ref,   // [B, LQ, NL, 2] f32
    _Float16*    __restrict__ outh,  // [B, LQ, NH, DH] f16
    int LQv)
{
    const int wave = threadIdx.x >> 5;
    const int lane = threadIdx.x & 31;
    const long long g = (long long)blockIdx.x * 8 + wave;  // (b*LQ+q)*NH + h
    const int h = (int)(g & 7);
    const long long bq = g >> 3;                           // b*LQ + q

    const int Hs_[4]    = {128, 64, 32, 16};
    const int Ws_[4]    = {128, 64, 32, 16};
    const int start_[4] = {0, 16384, 20480, 21504};

    const float* offp = off + bq * 256 + (size_t)h * 32;
    const float* awp  = aw  + bq * 128 + (size_t)h * 16;
    const float* refp = ref + bq * 8;
    const float* valb = val + (bq / LQv) * (long long)LQv * 256;  // batch base

    float acc = 0.0f;
#pragma unroll
    for (int l = 0; l < 4; ++l) {
        const int Hs = Hs_[l], Ws = Ws_[l], st = start_[l];
        const float rx = refp[l * 2 + 0];
        const float ry = refp[l * 2 + 1];
#pragma unroll
        for (int p = 0; p < 4; ++p) {
            const float ox = offp[l * 8 + p * 2 + 0];
            const float oy = offp[l * 8 + p * 2 + 1];
            const float a  = awp[l * 4 + p];

            const float x = (rx + ox / (float)Ws) * (float)Ws - 0.5f;
            const float y = (ry + oy / (float)Hs) * (float)Hs - 0.5f;
            const float xf = floorf(x), yf = floorf(y);
            const float lx = x - xf, ly = y - yf;
            const int x0 = (int)xf, y0 = (int)yf;

            const float w00 = (1.0f - lx) * (1.0f - ly);
            const float w10 = lx * (1.0f - ly);
            const float w01 = (1.0f - lx) * ly;
            const float w11 = lx * ly;

            float s = 0.0f;
#pragma unroll
            for (int c = 0; c < 4; ++c) {
                const int xi = x0 + (c & 1);
                const int yi = y0 + (c >> 1);
                const float w = (c == 0) ? w00 : (c == 1) ? w10 : (c == 2) ? w01 : w11;
                const bool inb = (xi >= 0) & (xi < Ws) & (yi >= 0) & (yi < Hs);
                int xc = xi < 0 ? 0 : (xi > Ws - 1 ? Ws - 1 : xi);
                int yc = yi < 0 ? 0 : (yi > Hs - 1 ? Hs - 1 : yi);
                const long long idx = st + (long long)yc * Ws + xc;
                const float vv = valb[(idx * 8 + h) * 32 + lane];
                s += inb ? w * vv : 0.0f;
            }
            acc += a * s;
        }
    }
    outh[g * 32 + lane] = (_Float16)acc;
}

// ---------------------------------------------------------------------------
// Host-side orchestration.
// ---------------------------------------------------------------------------
extern "C" void kernel_launch(void* const* d_in, const int* in_sizes, int n_in,
                              void* d_out, int out_size, void* d_ws, size_t ws_size,
                              hipStream_t stream)
{
    (void)in_sizes; (void)n_in; (void)out_size; (void)ws_size;

    const float* src    = (const float*)d_in[0];
    const float* pos    = (const float*)d_in[1];
    const float* ref    = (const float*)d_in[2];
    // d_in[3] spatial_shapes, d_in[4] level_start_index: hardcoded (fixed shapes)
    const float* g1     = (const float*)d_in[5];
    const float* be1    = (const float*)d_in[6];
    const float* w_off  = (const float*)d_in[7];
    const float* b_off  = (const float*)d_in[8];
    const float* w_attn = (const float*)d_in[9];
    const float* b_attn = (const float*)d_in[10];
    const float* w_val  = (const float*)d_in[11];
    const float* b_val  = (const float*)d_in[12];
    const float* w_out  = (const float*)d_in[13];
    const float* b_out  = (const float*)d_in[14];
    const float* g2     = (const float*)d_in[15];
    const float* be2    = (const float*)d_in[16];
    const float* w_fc1  = (const float*)d_in[17];
    const float* b_fc1  = (const float*)d_in[18];
    const float* w_fc2  = (const float*)d_in[19];
    const float* b_fc2  = (const float*)d_in[20];

    const int Bsz = 2, LQ = 21760;
    const int M = Bsz * LQ;               // 43520 (= 340 * 128)

    // ---- workspace carve-out (256-byte aligned), with lifetime aliasing ----
    char* ws = (char*)d_ws;
    size_t cur = 0;
    auto carve = [&](size_t bytes) -> char* {
        char* p = ws + cur;
        cur = (cur + bytes + 255) & ~(size_t)255;
        return p;
    };
    _Float16* wvalT  = (_Float16*)carve((size_t)256 * 256 * 2);
    _Float16* woffT  = (_Float16*)carve((size_t)256 * 256 * 2);
    _Float16* wattnT = (_Float16*)carve((size_t)128 * 256 * 2);
    _Float16* woutT  = (_Float16*)carve((size_t)256 * 256 * 2);
    _Float16* wfc1T  = (_Float16*)carve((size_t)1024 * 256 * 2);
    _Float16* wfc2T  = (_Float16*)carve((size_t)256 * 1024 * 2);
    _Float16* xh   = (_Float16*)carve((size_t)M * 256 * 2);   // LN1(src); dead after val GEMM
    _Float16* qh   = (_Float16*)carve((size_t)M * 256 * 2);   // LN1+pos; dead after off/attn GEMMs
    float*    valf = (float*)   carve((size_t)M * 256 * 4);   // dead after sampling
    float*    offf = (float*)   carve((size_t)M * 256 * 4);
    float*    awf  = (float*)   carve((size_t)M * 128 * 4);
    _Float16* hh   = (_Float16*)carve((size_t)M * 1024 * 2);
    _Float16* outh = xh;      // alias: sampled heads (f16), after xh is dead
    float*    src2 = valf;    // alias: attention residual output, after valf is dead
    _Float16* yh   = qh;      // alias: LN2 output, after qh is dead

    // ---- 1) weight convert+transpose (tiny; L2-resident afterwards) ----
    auto cvt = [&](const float* W, _Float16* Wt, int K, int N) {
        int total = K * N;
        convert_w_kernel<<<(total + 255) / 256, 256, 0, stream>>>(W, Wt, K, N);
    };
    cvt(w_val,  wvalT,  256, 256);
    cvt(w_off,  woffT,  256, 256);
    cvt(w_attn, wattnT, 256, 128);
    cvt(w_out,  woutT,  256, 256);
    cvt(w_fc1,  wfc1T,  256, 1024);
    cvt(w_fc2,  wfc2T,  1024, 256);

    auto gemm = [&](const _Float16* A, const _Float16* Wt, const float* bias,
                    const float* resid, float* Cf, _Float16* Ch,
                    int N, int K, int relu) {
        dim3 grid((unsigned)((M / 128) * (N / 64)));
        gemm_wmma_kernel<<<grid, 256, 0, stream>>>(A, Wt, bias, resid, Cf, Ch,
                                                   M, N, K, relu);
    };

    // ---- 2) LN1 -> xh (f16) and qh = LN1+pos (f16) ----
    ln_kernel<<<M / 8, 256, 0, stream>>>(src, pos, g1, be1, xh, qh, M);

    // ---- 3) projections via WMMA ----
    gemm(xh, wvalT,  b_val,  nullptr, valf, nullptr, 256, 256, 0);   // val  (f32)
    gemm(qh, woffT,  b_off,  nullptr, offf, nullptr, 256, 256, 0);   // off  (f32)
    gemm(qh, wattnT, b_attn, nullptr, awf,  nullptr, 128, 256, 0);   // attn logits

    // ---- 4) softmax over NL*NP per (b,q,h) ----
    softmax16_kernel<<<(M * 8 + 255) / 256, 256, 0, stream>>>(awf, M * 8);

    // ---- 5) deformable sampling -> f16 heads ----
    deform_sample_kernel<<<M, 256, 0, stream>>>(valf, offf, awf, ref, outh, LQ);

    // ---- 6) out projection + residual: src2 = src + outh @ w_out + b_out ----
    gemm(outh, woutT, b_out, src, src2, nullptr, 256, 256, 0);

    // ---- 7) LN2 -> yh (f16) ----
    ln_kernel<<<M / 8, 256, 0, stream>>>(src2, nullptr, g2, be2, yh, nullptr, M);

    // ---- 8) MLP: hh = relu(yh @ fc1 + b1);  out = src2 + hh @ fc2 + b2 ----
    gemm(yh, wfc1T, b_fc1, nullptr, nullptr, hh, 1024, 256, 1);
    gemm(hh, wfc2T, b_fc2, src2, (float*)d_out, nullptr, 256, 1024, 0);
}
